// SchNet_5162550690481
// MI455X (gfx1250) — matrix operand order repaired and checked by reading, not compiled, not run on hardware
//
#include <hip/hip_runtime.h>

#define N_ATOMS 512
#define FEAT    128
#define NKER    301
#define KPAD    320
#define LOG2C   0.69314718055994530942f

typedef __attribute__((ext_vector_type(16))) __bf16 v16bf;
typedef __attribute__((ext_vector_type(8)))  float  v8f;

union ABFrag { v16bf v; unsigned u[8]; uint4 q[2]; unsigned short s[16]; };

__device__ __forceinline__ unsigned short f2bf(float x) {
  unsigned u = __float_as_uint(x);
  unsigned r = u + 0x7FFFu + ((u >> 16) & 1u);   // round-to-nearest-even
  return (unsigned short)(r >> 16);
}

__device__ __forceinline__ float sspf(float x) {
  // softplus(x) - log(2), numerically stable
  return fmaxf(x, 0.f) + __logf(1.f + __expf(-fabsf(x))) - LOG2C;
}

// A-fragment: lane-row base, K-groups {0..7 | 16..23} (hi=0) / {8..15 | 24..31} (hi=1)
// -> two 16B-aligned b128 LDS loads
__device__ __forceinline__ void ld_fragA(ABFrag& f, const unsigned* row, int hi) {
  f.q[0] = *(const uint4*)(row + 4*hi);
  f.q[1] = *(const uint4*)(row + 8 + 4*hi);
}
// B-fragment: dwords {0..3 | 8..11} from base (base = col + 4*hi; K-groups as A) -> two b128 LDS loads
__device__ __forceinline__ void ld_fragB(ABFrag& f, const unsigned* base) {
  f.q[0] = *(const uint4*)(base);
  f.q[1] = *(const uint4*)(base + 8);
}

// ---------------------------------------------------------------------------
// Pairwise distances: d[i,j] = ||r_i - r_j||, 0 on the diagonal
// ---------------------------------------------------------------------------
__global__ void __launch_bounds__(256)
dist_kernel(const float* __restrict__ r, float* __restrict__ dmat)
{
  int idx = blockIdx.x * 256 + threadIdx.x;       // 0 .. 512*512-1
  int i = idx >> 9, j = idx & 511;
  float dx = r[i*3+0] - r[j*3+0];
  float dy = r[i*3+1] - r[j*3+1];
  float dz = r[i*3+2] - r[j*3+2];
  float d2 = dx*dx + dy*dy + dz*dz;
  dmat[idx] = d2 > 0.f ? sqrtf(d2) : 0.f;
}

// ---------------------------------------------------------------------------
// Embedding gather: h = emb[x]
// ---------------------------------------------------------------------------
__global__ void __launch_bounds__(256)
embed_kernel(const int* __restrict__ x, const float* __restrict__ emb,
             float* __restrict__ h)
{
  int idx = blockIdx.x * 256 + threadIdx.x;       // 512*128
  int i = idx >> 7, c = idx & 127;
  h[idx] = emb[x[i]*FEAT + c];
}

__global__ void __launch_bounds__(256)
zero_kernel(float* __restrict__ p, int n)
{
  int i = blockIdx.x * blockDim.x + threadIdx.x;
  if (i < n) p[i] = 0.f;
}

// ---------------------------------------------------------------------------
// Small dense layer: Y[i,c] = (addTo?) + act( sum_k X[i,k]*W[k,c] + B[c] )
// ---------------------------------------------------------------------------
__global__ void __launch_bounds__(256)
gemm_small(const float* __restrict__ X, const float* __restrict__ W,
           const float* __restrict__ B, const float* __restrict__ addTo,
           float* __restrict__ Y, int Kdim, int Cout, int act)
{
  int idx = blockIdx.x * 256 + threadIdx.x;
  int i = idx / Cout, c = idx % Cout;
  float s = B[c];
  for (int k = 0; k < Kdim; ++k) s += X[i*Kdim + k] * W[k*Cout + c];
  if (act) s = sspf(s);
  if (addTo) s += addTo[idx];
  Y[idx] = s;
}

// ---------------------------------------------------------------------------
// Fused continuous-filter conv layer (the ~90-GFLOP core).
// Block = (16 i) x (16 j) pair tile -> 256 GEMM rows.  8 waves.
// GEMM1: rbf(d)[256 x 320] @ W1t  streamed in K-chunks of 32 (bf16 WMMA)
// GEMM2: ssp(.)[256 x 128] @ W2t  (bf16 WMMA)
// conv:  lRed[il,c] += o[j,c] * ssp(.)  via ds_add_f32, then 1 global atomic
// ---------------------------------------------------------------------------
__global__ void __launch_bounds__(256)
schnet_conv_kernel(const float* __restrict__ dmat,
                   const float* __restrict__ o,     // [512,128] f32
                   const float* __restrict__ w1,    // [301,128] f32 (layer slice)
                   const float* __restrict__ b1,    // [128]
                   const float* __restrict__ w2,    // [128,128]
                   const float* __restrict__ b2,    // [128]
                   float* __restrict__ cbuf)        // [512,128] accum
{
  extern __shared__ char smem[];
  unsigned short* lA   = (unsigned short*)smem;     // [256][32]  bf16 RBF chunk
  unsigned short* lBt  = lA  + 256*32;              // [128][32]  bf16 W1^T chunk
  unsigned short* lS   = lBt + 128*32;              // [256][128] bf16 intermediate
  unsigned short* lW2t = lS  + 256*128;             // [128][128] bf16 W2^T
  float* lO   = (float*)(lW2t + 128*128);           // [16][128]  o tile (rows j)
  float* lRed = lO + 16*128;                        // [16][128]  conv reduction

  const unsigned* lA32  = (const unsigned*)lA;
  const unsigned* lBt32 = (const unsigned*)lBt;
  const unsigned* lS32  = (const unsigned*)lS;
  const unsigned* lW32  = (const unsigned*)lW2t;

  const int tid  = threadIdx.x;
  const int wave = tid >> 5;
  const int lane = tid & 31;
  const int lh   = lane & 15;
  const int hi   = lane >> 4;
  const int bi   = blockIdx.x;
  const int bj   = blockIdx.y;

  // this thread's pair distance (pair p == tid : i-local = p>>4, j-local = p&15)
  const int   myI = bi*16 + (tid >> 4);
  const int   myJ = bj*16 + (tid & 15);
  const float dp  = dmat[myI*N_ATOMS + myJ];

  // biases for this lane's 8 column slots (col = nt*16 + lh)
  float bia1[8], bia2[8];
  #pragma unroll
  for (int nt = 0; nt < 8; ++nt) {
    bia1[nt] = b1[nt*16 + lh];
    bia2[nt] = b2[nt*16 + lh];
  }

  // stage o tile + zero the reduction buffer
  #pragma unroll
  for (int q = 0; q < 8; ++q) {
    int n = tid*8 + q;                              // 0..2047
    lO[n]   = o[(bj*16 + (n >> 7))*FEAT + (n & 127)];
    lRed[n] = 0.f;
  }
  // stage transposed W2 as bf16: lW2t[c][k]
  #pragma unroll
  for (int q = 0; q < 64; ++q) {
    int n = tid*64 + q;                             // 0..16383
    int c = n >> 7, k = n & 127;
    lW2t[c*128 + k] = f2bf(w2[k*FEAT + c]);
  }

  v8f acc[16];                                      // [mt*8+nt]
  #pragma unroll
  for (int t = 0; t < 16; ++t)
    #pragma unroll
    for (int rr = 0; rr < 8; ++rr) acc[t][rr] = 0.f;

  const int mrow0 = wave * 32;                      // wave's first pair row
  const unsigned* aRow0 = lA32 + (mrow0 + lh) * 16;
  const unsigned* aRow1 = lA32 + (mrow0 + 16 + lh) * 16;

  // ---- GEMM1: streamed over K (301 padded to 320, chunks of 32) ----
  for (int ks = 0; ks < KPAD/32; ++ks) {
    const int kbase = ks * 32;
    // A chunk: Gaussian RBF expansion, on the fly, bf16
    #pragma unroll
    for (int kk = 0; kk < 32; ++kk) {
      int kg = kbase + kk;
      float v = 0.f;
      if (kg < NKER) {
        float t0 = dp - 0.1f * (float)kg;           // centers = linspace(0,30,301)
        v = __expf(-10.f * t0 * t0);
      }
      lA[tid*32 + kk] = f2bf(v);
    }
    // B chunk transposed: lBt[c][k] = W1[kbase+k][c] (zero rows past 301)
    #pragma unroll
    for (int q = 0; q < 16; ++q) {
      int n = tid*16 + q;                           // 0..4095
      int c = n >> 5, k = n & 31;
      int row = kbase + k;
      lBt[c*32 + k] = f2bf(row < NKER ? w1[row*FEAT + c] : 0.f);
    }
    // prefetch next W1 chunk while this one is consumed (global_prefetch_b8)
    if (ks + 1 < KPAD/32 && kbase + 32 < NKER)
      __builtin_prefetch(&w1[(kbase + 32) * FEAT + (tid & 127)], 0, 1);
    __syncthreads();

    ABFrag a0, a1;
    ld_fragA(a0, aRow0, hi);
    ld_fragA(a1, aRow1, hi);
    #pragma unroll
    for (int nt = 0; nt < 8; ++nt) {
      ABFrag b;
      ld_fragB(b, lBt32 + (nt*16 + lh)*16 + 4*hi);
      acc[nt]     = __builtin_amdgcn_wmma_f32_16x16x32_bf16(false, a0.v, false, b.v, (short)0, acc[nt],     false, false);
      acc[8 + nt] = __builtin_amdgcn_wmma_f32_16x16x32_bf16(false, a1.v, false, b.v, (short)0, acc[8 + nt], false, false);
    }
    __syncthreads();
  }

  // bias + ShiftedSoftplus -> bf16 intermediate in lS; reset accumulators
  #pragma unroll
  for (int mt = 0; mt < 2; ++mt)
    #pragma unroll
    for (int nt = 0; nt < 8; ++nt)
      #pragma unroll
      for (int rr = 0; rr < 8; ++rr) {
        int p   = mrow0 + mt*16 + rr + hi*8;
        int col = nt*16 + lh;
        lS[p*FEAT + col] = f2bf(sspf(acc[mt*8+nt][rr] + bia1[nt]));
        acc[mt*8+nt][rr] = 0.f;
      }
  __syncthreads();

  // ---- GEMM2: [256 x 128] @ W2 [128 x 128], 4 K-steps ----
  const unsigned* sRow0 = lS32 + (mrow0 + lh) * 64;
  const unsigned* sRow1 = lS32 + (mrow0 + 16 + lh) * 64;
  #pragma unroll
  for (int ks = 0; ks < 4; ++ks) {
    ABFrag a0, a1;
    ld_fragA(a0, sRow0 + ks*16, hi);
    ld_fragA(a1, sRow1 + ks*16, hi);
    #pragma unroll
    for (int nt = 0; nt < 8; ++nt) {
      ABFrag b;
      ld_fragB(b, lW32 + (nt*16 + lh)*64 + ks*16 + 4*hi);
      acc[nt]     = __builtin_amdgcn_wmma_f32_16x16x32_bf16(false, a0.v, false, b.v, (short)0, acc[nt],     false, false);
      acc[8 + nt] = __builtin_amdgcn_wmma_f32_16x16x32_bf16(false, a1.v, false, b.v, (short)0, acc[8 + nt], false, false);
    }
  }

  // bias + ssp -> filter w(i,j,col); conv reduce over j-local in LDS
  #pragma unroll
  for (int mt = 0; mt < 2; ++mt)
    #pragma unroll
    for (int nt = 0; nt < 8; ++nt)
      #pragma unroll
      for (int rr = 0; rr < 8; ++rr) {
        int p   = mrow0 + mt*16 + rr + hi*8;
        int col = nt*16 + lh;
        int il = p >> 4, jl = p & 15;
        float wv = sspf(acc[mt*8+nt][rr] + bia2[nt]);
        atomicAdd(&lRed[il*FEAT + col], lO[jl*FEAT + col] * wv);
      }
  __syncthreads();

  // one global atomic per (i-local, col) per block
  #pragma unroll
  for (int q = 0; q < 8; ++q) {
    int n = tid*8 + q;
    atomicAdd(&cbuf[(bi*16 + (n >> 7))*FEAT + (n & 127)], lRed[n]);
  }
}

// ---------------------------------------------------------------------------
// Output head: out[i] = ssp(h @ W1 + b1) @ W2 + b2
// ---------------------------------------------------------------------------
__global__ void __launch_bounds__(256)
head_kernel(const float* __restrict__ h, const float* __restrict__ W1,
            const float* __restrict__ B1, const float* __restrict__ W2,
            const float* __restrict__ B2, float* __restrict__ out)
{
  int i = blockIdx.x * 256 + threadIdx.x;
  if (i >= N_ATOMS) return;
  float accv = B2[0];
  for (int u = 0; u < 32; ++u) {
    float s = B1[u];
    for (int k = 0; k < FEAT; ++k) s += h[i*FEAT + k] * W1[k*32 + u];
    accv += sspf(s) * W2[u];
  }
  out[i] = accv;
}

// ---------------------------------------------------------------------------
extern "C" void kernel_launch(void* const* d_in, const int* in_sizes, int n_in,
                              void* d_out, int out_size, void* d_ws, size_t ws_size,
                              hipStream_t stream)
{
  (void)in_sizes; (void)n_in; (void)out_size; (void)ws_size;

  const int*   x      = (const int*)  d_in[0];
  const float* r      = (const float*)d_in[1];
  const float* emb    = (const float*)d_in[2];
  const float* aw_W   = (const float*)d_in[3];
  const float* aw_b   = (const float*)d_in[4];
  const float* cf_W1  = (const float*)d_in[5];
  const float* cf_b1  = (const float*)d_in[6];
  const float* cf_W2  = (const float*)d_in[7];
  const float* cf_b2  = (const float*)d_in[8];
  const float* n_W1   = (const float*)d_in[9];
  const float* n_b1   = (const float*)d_in[10];
  const float* n_W2   = (const float*)d_in[11];
  const float* n_b2   = (const float*)d_in[12];
  const float* out_W1 = (const float*)d_in[13];
  const float* out_b1 = (const float*)d_in[14];
  const float* out_W2 = (const float*)d_in[15];
  const float* out_b2 = (const float*)d_in[16];
  float* out = (float*)d_out;

  float* dmat = (float*)d_ws;                     // 512*512
  float* h    = dmat + N_ATOMS*N_ATOMS;           // 512*128
  float* o    = h    + N_ATOMS*FEAT;              // 512*128
  float* cbuf = o    + N_ATOMS*FEAT;              // 512*128
  float* t1   = cbuf + N_ATOMS*FEAT;              // 512*128

  dist_kernel <<<(N_ATOMS*N_ATOMS)/256, 256, 0, stream>>>(r, dmat);
  embed_kernel<<<(N_ATOMS*FEAT)/256,    256, 0, stream>>>(x, emb, h);

  const size_t ldsBytes =
      (size_t)(256*32 + 128*32 + 256*128 + 128*128) * sizeof(unsigned short)
    + (size_t)(2*16*128) * sizeof(float);

  for (int l = 0; l < 3; ++l) {
    gemm_small<<<(N_ATOMS*FEAT)/256, 256, 0, stream>>>(
        h, aw_W + l*FEAT*FEAT, aw_b + l*FEAT, nullptr, o, FEAT, FEAT, 0);
    zero_kernel<<<(N_ATOMS*FEAT)/256, 256, 0, stream>>>(cbuf, N_ATOMS*FEAT);
    schnet_conv_kernel<<<dim3(32, 32), 256, ldsBytes, stream>>>(
        dmat, o, cf_W1 + l*NKER*FEAT, cf_b1 + l*FEAT,
        cf_W2 + l*FEAT*FEAT, cf_b2 + l*FEAT, cbuf);
    gemm_small<<<(N_ATOMS*FEAT)/256, 256, 0, stream>>>(
        cbuf, n_W1 + l*FEAT*FEAT, n_b1 + l*FEAT, nullptr, t1, FEAT, FEAT, 1);
    gemm_small<<<(N_ATOMS*FEAT)/256, 256, 0, stream>>>(
        t1, n_W2 + l*FEAT*FEAT, n_b2 + l*FEAT, h, h, FEAT, FEAT, 0);
  }

  head_kernel<<<(N_ATOMS + 255)/256, 256, 0, stream>>>(
      h, out_W1, out_b1, out_W2, out_b2, out);
}
